// SelfAttention_1_9277129359983
// MI455X (gfx1250) — compile-verified, matrix operand order checked by reading, not run on hardware
//
#include <hip/hip_runtime.h>
#include <hip/hip_bf16.h>

// ---------------------------------------------------------------------------
// SelfAttention_1 for MI455X (gfx1250, wave32, WMMA).
//
// Shapes: x [4, 256, 64, 64] fp32.  Per batch:
//   K  = row-normalized (x.reshape(256,4096).T + eps)            [4096, 256]
//   P  = K @ Y   where Y = 3x3 zero-padded box-sum of Xpad       [4096, 4356]
//        (pooling commutes with the GEMM -> pool the small side once)
//   A  = softmax(P, axis=0)  -> written straight into d_out att region
//        in (u,v,l) order (column of the softmax axis contiguous)
//   out= K^T @ A_crop  ==  (A'_rows @ K)^T with A' rows = att columns
//
// d_out = [ out: 4*256*4096 fp32 | att: 4*4096*4356 fp32 ]
//
// Roofline: ~71 GFLOP of GEMM (trivial for bf16 WMMA) vs ~1.2 GB of HBM
// traffic dominated by the 285 MB softmax matrix -> bandwidth bound; all
// passes over that matrix use contiguous, b128-vectorized access.
// ---------------------------------------------------------------------------

typedef __bf16 bf16;
typedef __attribute__((ext_vector_type(16))) __bf16 v16bf;
typedef __attribute__((ext_vector_type(8)))  float  v8f;

#define BZ     4
#define NCH    256
#define LDIM   4096        // 64*64 softmax axis
#define MCOLS  4356        // 66*66 padded spatial columns
#define NPAD   4480        // MCOLS rounded up to 128 (GEMM1 N tiles)
#define EPSF   1e-7f

static constexpr size_t ATT_BSTRIDE = (size_t)LDIM * MCOLS;   // 17,842,176
static constexpr size_t OUT_BSTRIDE = (size_t)NCH * LDIM;     // 1,048,576
static constexpr size_t OUT_ELEMS   = (size_t)BZ * OUT_BSTRIDE;

// --------------------------------------------------------------------------
// K normalization: Kbf [b][l][c] and KTbf [b][c][l] in bf16.
// One thread per l; loops over c with coalesced loads (l contiguous in x).
// --------------------------------------------------------------------------
__global__ __launch_bounds__(256) void prep_norm(const float* __restrict__ x,
                                                 bf16* __restrict__ Kbf,
                                                 bf16* __restrict__ KTbf) {
  const int b = blockIdx.y;
  const int l = blockIdx.x * 256 + threadIdx.x;
  const float* xb = x + (size_t)b * NCH * LDIM;
  float ss = 0.f;
  for (int c = 0; c < NCH; ++c) {
    float v = xb[(size_t)c * LDIM + l] + EPSF;
    ss += v * v;
  }
  const float rn = rsqrtf(ss);
  bf16* kb = Kbf  + (size_t)b * LDIM * NCH;
  bf16* kt = KTbf + (size_t)b * NCH * LDIM;
  for (int c = 0; c < NCH; ++c) {
    float v = (xb[(size_t)c * LDIM + l] + EPSF) * rn;
    bf16 h = (bf16)v;
    kb[(size_t)l * NCH + c] = h;   // row-major K  (GEMM1 A operand)
    kt[(size_t)c * LDIM + l] = h;  // row-major K^T (GEMM2 B operand)
  }
}

// --------------------------------------------------------------------------
// Pooled, transposed B operand for GEMM1: YT [b][j][c] bf16, j in [0,NPAD),
// YT[j][c] = sum over 3x3 window (both paddings folded) of x[c].
// Block = one (b, c); threads stride over j -> spatially coalesced reads.
// --------------------------------------------------------------------------
__global__ __launch_bounds__(256) void prep_y(const float* __restrict__ x,
                                              bf16* __restrict__ YT) {
  const int b = blockIdx.y;
  const int c = blockIdx.x;
  const float* xc = x + ((size_t)b * NCH + c) * LDIM;
  bf16* yb = YT + (size_t)b * NPAD * NCH;
  for (int j = threadIdx.x; j < NPAD; j += 256) {
    float s = 0.f;
    if (j < MCOLS) {
      const int u = j / 66, v = j % 66;
      #pragma unroll
      for (int dr = 0; dr < 3; ++dr) {
        const int r = u - 2 + dr;
        if (r < 0 || r >= 64) continue;
        #pragma unroll
        for (int ds = 0; ds < 3; ++ds) {
          const int sc = v - 2 + ds;
          if (sc < 0 || sc >= 64) continue;
          s += xc[r * 64 + sc];
        }
      }
    }
    yb[(size_t)j * NCH + c] = (bf16)s;  // zero rows pad j in [4356,4480)
  }
}

// --------------------------------------------------------------------------
// GEMM1: P[l, j] = K[l, :] . YT[j, :]   (M=4096, N=4480, K=256, bf16 WMMA)
// Block 256 thr = 8 waves as 4(M) x 2(N); wave tile 32x64 = 2x4 subtiles.
// A frag: lane&15 = M row, lane>>4 = K half -> contiguous 32B per lane.
// B frag: lane&15 = N col, lane>>4 = K half -> contiguous 32B (YT is B^T).
// Store transposed into d_out att region: att[j*4096 + l]; each lane's 8
// accumulator elements are consecutive l -> one aligned v8f (b128x2) store.
// --------------------------------------------------------------------------
__global__ __launch_bounds__(256) void gemm1_scores(const bf16* __restrict__ Kbf,
                                                    const bf16* __restrict__ YT,
                                                    float* __restrict__ att) {
  const int b    = blockIdx.z;
  const int tid  = threadIdx.x;
  const int wave = tid >> 5, lane = tid & 31;
  const int wm   = wave & 3, wn = wave >> 2;
  const int lrow = lane & 15, kh = lane >> 4;
  const bf16* A = Kbf + (size_t)b * LDIM * NCH;
  const bf16* B = YT  + (size_t)b * NPAD * NCH;
  float* P = att + (size_t)b * ATT_BSTRIDE;
  const int m0 = blockIdx.y * 128 + wm * 32;  // l base
  const int n0 = blockIdx.x * 128 + wn * 64;  // j base
  v8f acc[2][4] = {};
  for (int k0 = 0; k0 < NCH; k0 += 32) {
    v16bf af[2], bfr[4];
    #pragma unroll
    for (int i = 0; i < 2; ++i)
      af[i] = *(const v16bf*)(A + (size_t)(m0 + i * 16 + lrow) * NCH + k0 + kh * 16);
    #pragma unroll
    for (int n = 0; n < 4; ++n)
      bfr[n] = *(const v16bf*)(B + (size_t)(n0 + n * 16 + lrow) * NCH + k0 + kh * 16);
    #pragma unroll
    for (int i = 0; i < 2; ++i)
      #pragma unroll
      for (int n = 0; n < 4; ++n)
        acc[i][n] = __builtin_amdgcn_wmma_f32_16x16x32_bf16(
            false, af[i], false, bfr[n], (short)0, acc[i][n], false, false);
  }
  // C layout (ISA 7.12.2): acc[r] holds M = r + 8*(lane>>4), N = lane&15.
  #pragma unroll
  for (int i = 0; i < 2; ++i) {
    #pragma unroll
    for (int n = 0; n < 4; ++n) {
      const int j = n0 + n * 16 + lrow;
      if (j < MCOLS) {
        const int l0 = m0 + i * 16 + kh * 8;          // 8-aligned
        *(v8f*)(P + (size_t)j * LDIM + l0) = acc[i][n];
      }
    }
  }
}

// --------------------------------------------------------------------------
// In-place softmax over the L axis: one block per (b, j) column of 4096
// contiguous fp32 in the att region.  Each thread owns two aligned v8f
// chunks (b128-vectorized read/modify/write of the 285 MB matrix).
// --------------------------------------------------------------------------
__global__ __launch_bounds__(256) void softmax_col(float* __restrict__ att) {
  __shared__ float red[256];
  const int b = blockIdx.y, j = blockIdx.x, t = threadIdx.x;
  float* col = att + (size_t)b * ATT_BSTRIDE + (size_t)j * LDIM;
  v8f v0 = *(const v8f*)(col + t * 8);
  v8f v1 = *(const v8f*)(col + 2048 + t * 8);
  float mx = -1e30f;
  #pragma unroll
  for (int i = 0; i < 8; ++i) mx = fmaxf(mx, fmaxf(v0[i], v1[i]));
  red[t] = mx; __syncthreads();
  for (int s = 128; s > 0; s >>= 1) {
    if (t < s) red[t] = fmaxf(red[t], red[t + s]);
    __syncthreads();
  }
  mx = red[0]; __syncthreads();
  float sum = 0.f;
  #pragma unroll
  for (int i = 0; i < 8; ++i) {
    v0[i] = __expf(v0[i] - mx); sum += v0[i];
    v1[i] = __expf(v1[i] - mx); sum += v1[i];
  }
  red[t] = sum; __syncthreads();
  for (int s = 128; s > 0; s >>= 1) {
    if (t < s) red[t] += red[t + s];
    __syncthreads();
  }
  const float inv = 1.0f / red[0];
  #pragma unroll
  for (int i = 0; i < 8; ++i) { v0[i] *= inv; v1[i] *= inv; }
  *(v8f*)(col + t * 8) = v0;
  *(v8f*)(col + 2048 + t * 8) = v1;
}

// --------------------------------------------------------------------------
// GEMM2: out^T[p, c] = A'[p, :] . K[:, c]  (M=4096, N=256, K=4096)
// A' row p = att column at ((p/64+1)*66 + (p%64+1))*4096, fp32 -> bf16 cvt.
// B = KTbf (K^T row-major) -> contiguous B fragments.
// Block tile 64(M) x 256(N): full N per block so A' is read exactly once.
// Store transposed: out[c*4096 + p], 8 consecutive p -> aligned v8f store.
// --------------------------------------------------------------------------
__global__ __launch_bounds__(256) void gemm2_out(const bf16*  __restrict__ KTbf,
                                                 const float* __restrict__ att,
                                                 float* __restrict__ out) {
  const int b    = blockIdx.z;
  const int tid  = threadIdx.x;
  const int wave = tid >> 5, lane = tid & 31;
  const int wm   = wave & 1, wn = wave >> 1;       // 2(M) x 4(N) waves
  const int lrow = lane & 15, kh = lane >> 4;
  const float* Ab = att  + (size_t)b * ATT_BSTRIDE;
  const bf16*  Bb = KTbf + (size_t)b * NCH * LDIM;
  float* ob = out + (size_t)b * OUT_BSTRIDE;
  const int m0 = blockIdx.x * 64 + wm * 32;        // p base
  const int n0 = wn * 64;                          // c base
  size_t arow[2];
  #pragma unroll
  for (int i = 0; i < 2; ++i) {
    const int p = m0 + i * 16 + lrow;
    arow[i] = ((size_t)((p >> 6) + 1) * 66 + (size_t)((p & 63) + 1)) * LDIM;
  }
  v8f acc[2][4] = {};
  for (int k0 = 0; k0 < LDIM; k0 += 32) {
    v16bf af[2], bfr[4];
    #pragma unroll
    for (int i = 0; i < 2; ++i) {
      const float* pr = Ab + arow[i] + k0 + kh * 16;
      v8f x0 = *(const v8f*)(pr);
      v8f x1 = *(const v8f*)(pr + 8);
      v16bf tv;
      #pragma unroll
      for (int e = 0; e < 8; ++e) { tv[e] = (bf16)x0[e]; tv[e + 8] = (bf16)x1[e]; }
      af[i] = tv;
    }
    #pragma unroll
    for (int n = 0; n < 4; ++n)
      bfr[n] = *(const v16bf*)(Bb + (size_t)(n0 + n * 16 + lrow) * LDIM + k0 + kh * 16);
    #pragma unroll
    for (int i = 0; i < 2; ++i)
      #pragma unroll
      for (int n = 0; n < 4; ++n)
        acc[i][n] = __builtin_amdgcn_wmma_f32_16x16x32_bf16(
            false, af[i], false, bfr[n], (short)0, acc[i][n], false, false);
  }
  #pragma unroll
  for (int i = 0; i < 2; ++i)
    #pragma unroll
    for (int n = 0; n < 4; ++n) {
      const int c  = n0 + n * 16 + lrow;
      const int p0 = m0 + i * 16 + kh * 8;           // 8-aligned
      *(v8f*)(ob + (size_t)c * LDIM + p0) = acc[i][n];
    }
}

// --------------------------------------------------------------------------
extern "C" void kernel_launch(void* const* d_in, const int* in_sizes, int n_in,
                              void* d_out, int out_size, void* d_ws, size_t ws_size,
                              hipStream_t stream) {
  (void)in_sizes; (void)n_in; (void)out_size; (void)ws_size;
  const float* x = (const float*)d_in[0];
  float* out = (float*)d_out;
  float* att = out + OUT_ELEMS;            // att region of the output

  // Workspace: Kbf (8 MB) | KTbf (8 MB) | YT (8.75 MB)  ~ 26 MB total
  char* ws = (char*)d_ws;
  bf16* Kbf  = (bf16*)(ws);
  bf16* KTbf = (bf16*)(ws + (size_t)BZ * LDIM * NCH * sizeof(bf16));
  bf16* YT   = (bf16*)(ws + (size_t)2 * BZ * LDIM * NCH * sizeof(bf16));

  prep_norm  <<<dim3(LDIM / 256, BZ),       256, 0, stream>>>(x, Kbf, KTbf);
  prep_y     <<<dim3(NCH, BZ),              256, 0, stream>>>(x, YT);
  gemm1_scores<<<dim3(NPAD / 128, LDIM / 128, BZ), 256, 0, stream>>>(Kbf, YT, att);
  softmax_col<<<dim3(MCOLS, BZ),            256, 0, stream>>>(att);
  gemm2_out  <<<dim3(LDIM / 64, 1, BZ),     256, 0, stream>>>(KTbf, att, out);
}
